// Decoder_17197049053348
// MI455X (gfx1250) — matrix operand order repaired and checked by reading, not compiled, not run on hardware
//
#include <hip/hip_runtime.h>

// Problem constants (from reference)
constexpr int E_   = 512;
constexpr int D_   = 1024;
constexpr int B_   = 64;      // batch == M for every GEMM here
constexpr int T_   = 128;
constexpr int S_   = 128;
constexpr int G4_  = 4 * D_;  // 4096 gate width

#define DEV __device__ __forceinline__

typedef __attribute__((ext_vector_type(16))) __bf16 v16bf;
typedef __attribute__((ext_vector_type(8)))  float  v8f;

DEV unsigned short f2bf(float f) {
  unsigned int u = __float_as_uint(f);
  u += 0x7FFFu + ((u >> 16) & 1u);           // round-to-nearest-even
  return (unsigned short)(u >> 16);
}
DEV float sigm(float x) { return 1.0f / (1.0f + expf(-x)); }

DEV v8f wmma_bf16(uint4 a0, uint4 a1, uint4 b0, uint4 b1, v8f acc) {
  union { uint4 q[2]; v16bf v; } a, b;
  a.q[0] = a0; a.q[1] = a1;
  b.q[0] = b0; b.q[1] = b1;
  return __builtin_amdgcn_wmma_f32_16x16x32_bf16(
      false, a.v, false, b.v, (short)0, acc, false, false);
}

// ---------------------------------------------------------------------------
// WMMA bf16 GEMM, M fixed to 64: each wave owns a full 64x16 column strip so
// the weight (B) fragment is loaded once and reused by 4 WMMAs -> 4x less L2
// weight traffic. Explicitly software-pipelined: loads for K-step k+32 are
// issued before the WMMAs consuming K-step k, so each WMMA group only waits
// on loads from one iteration back (partial s_wait_loadcnt, load/math overlap).
// A: activations, row-major [64, lda] (K contiguous)  -- bf16 (ushort)
// B: weights,     row-major [N, ldb]  (K contiguous)  -- bf16 (ushort)
// Fragment layouts per CDNA5 ISA 7.12.2 (16-bit A 16x32, B 32x16, f32 C/D).
// ---------------------------------------------------------------------------
DEV void gemm_seg4(v8f acc[4], const unsigned short* __restrict__ A, int lda,
                   const unsigned short* __restrict__ Bw, int ldb, int K,
                   int n0, int lane)
{
  const int kh = lane >> 4;        // lane half selects K sub-block
  const int r  = lane & 15;        // A row (M) / B row (N)
  const unsigned short* Ap = A  + (size_t)r * lda + kh * 8;
  const unsigned short* Bp = Bw + (size_t)(n0 + r) * ldb + kh * 8;

  // prologue: fragments for k = 0
  uint4 bq0 = *(const uint4*)(Bp);            // K = kh*8 + 0..7, 16 + kh*8 + 0..7
  uint4 bq1 = *(const uint4*)(Bp + 16);
  uint4 aq0[4], aq1[4];
#pragma unroll
  for (int mt = 0; mt < 4; ++mt) {
    const unsigned short* Am = Ap + (size_t)(mt * 16) * lda;
    aq0[mt] = *(const uint4*)(Am);            // K = kh*8 + 0..7
    aq1[mt] = *(const uint4*)(Am + 16);       // K = 16 + kh*8 + 0..7
  }

  for (int k = 32; k < K; k += 32) {
    // issue next-iteration loads first (in flight during this WMMA group)
    uint4 nb0 = *(const uint4*)(Bp + k);
    uint4 nb1 = *(const uint4*)(Bp + k + 16);
    uint4 na0[4], na1[4];
#pragma unroll
    for (int mt = 0; mt < 4; ++mt) {
      const unsigned short* Am = Ap + (size_t)(mt * 16) * lda;
      na0[mt] = *(const uint4*)(Am + k);
      na1[mt] = *(const uint4*)(Am + k + 16);
    }
#pragma unroll
    for (int mt = 0; mt < 4; ++mt)
      acc[mt] = wmma_bf16(aq0[mt], aq1[mt], bq0, bq1, acc[mt]);
    bq0 = nb0; bq1 = nb1;
#pragma unroll
    for (int mt = 0; mt < 4; ++mt) { aq0[mt] = na0[mt]; aq1[mt] = na1[mt]; }
  }
  // epilogue: last K-step
#pragma unroll
  for (int mt = 0; mt < 4; ++mt)
    acc[mt] = wmma_bf16(aq0[mt], aq1[mt], bq0, bq1, acc[mt]);
}

__global__ void gemm_bf16_wmma(
    float* __restrict__ C, int ldc,
    unsigned short* __restrict__ Cbf, int ldcbf,
    const unsigned short* __restrict__ A1, int lda1,
    const unsigned short* __restrict__ B1, int ldb1, int K1,
    const unsigned short* __restrict__ A2, int lda2,
    const unsigned short* __restrict__ B2, int ldb2, int K2,
    const unsigned short* __restrict__ A3, int lda3,
    const unsigned short* __restrict__ B3, int ldb3, int K3,
    const float* __restrict__ bias1, const float* __restrict__ bias2,
    int act_tanh)
{
  const int lane = threadIdx.x & 31;
  const int wave = threadIdx.x >> 5;
  const int n0 = (blockIdx.x * 4 + wave) << 4;  // blockDim.x == 128 (4 waves)

  v8f acc[4];
#pragma unroll
  for (int mt = 0; mt < 4; ++mt) acc[mt] = (v8f){0.f,0.f,0.f,0.f,0.f,0.f,0.f,0.f};

  gemm_seg4(acc, A1, lda1, B1, ldb1, K1, n0, lane);
  if (K2 > 0) gemm_seg4(acc, A2, lda2, B2, ldb2, K2, n0, lane);
  if (K3 > 0) gemm_seg4(acc, A3, lda3, B3, ldb3, K3, n0, lane);

  // C/D layout: n = lane&15 ; m = (lane>>4)*8 + r  (per 16x16 tile)
  const int n  = n0 + (lane & 15);
  const int mb = (lane >> 4) << 3;
  float bias = 0.f;
  if (bias1) bias += bias1[n];
  if (bias2) bias += bias2[n];
#pragma unroll
  for (int mt = 0; mt < 4; ++mt) {
#pragma unroll
    for (int r = 0; r < 8; ++r) {
      float v = acc[mt][r] + bias;
      if (act_tanh) v = tanhf(v);
      const int row = mt * 16 + mb + r;
      C[(size_t)row * ldc + n] = v;
      if (Cbf) Cbf[(size_t)row * ldcbf + n] = f2bf(v);
    }
  }
}

// ---------------------------------------------------------------------------
// Pointwise kernels
// ---------------------------------------------------------------------------
__global__ void conv_f32_bf16(const float* __restrict__ s,
                              unsigned short* __restrict__ d, int n)
{
  int i = blockIdx.x * blockDim.x + threadIdx.x;
  if (i < n) d[i] = f2bf(s[i]);
}

// emb_bf[(t*B+b)*E + e] = bf16(word_lut[input[b*T+t]*E + e])
__global__ void embed_bf16(const int* __restrict__ input,
                           const float* __restrict__ lut,
                           unsigned short* __restrict__ emb)
{
  int idx = blockIdx.x * blockDim.x + threadIdx.x;  // T*B*E
  int e  = idx & (E_ - 1);
  int tb = idx >> 9;            // / E_
  int b  = tb & (B_ - 1);
  int t  = tb >> 6;             // / B_
  int tok = input[b * T_ + t];
  emb[idx] = f2bf(lut[(size_t)tok * E_ + e]);
}

// gate order i,f,g,o
__global__ void lstm_pointwise(const float* __restrict__ gates,
                               float* __restrict__ c, float* __restrict__ h,
                               unsigned short* __restrict__ hbf)
{
  int idx = blockIdx.x * blockDim.x + threadIdx.x;  // B*D
  int b = idx >> 10, d = idx & (D_ - 1);
  const float* g = gates + (size_t)b * G4_;
  float gi = g[d], gf = g[D_ + d], gg = g[2 * D_ + d], go = g[3 * D_ + d];
  float cn = sigm(gf) * c[idx] + sigm(gi) * tanhf(gg);
  float hn = sigm(go) * tanhf(cn);
  c[idx] = cn; h[idx] = hn; hbf[idx] = f2bf(hn);
}

// scores[b,s] = dot(context[s,b,:], q[b,:]) ; one wave per (b,s)
__global__ void attn_scores(const float* __restrict__ q,
                            const float* __restrict__ context,
                            float* __restrict__ scores)
{
  int wid  = (blockIdx.x * blockDim.x + threadIdx.x) >> 5;
  int lane = threadIdx.x & 31;
  int b = wid >> 7, s = wid & (S_ - 1);
  const float* cp = context + (size_t)s * B_ * D_ + (size_t)b * D_;
  const float* qp = q + (size_t)b * D_;
  float sum = 0.f;
  for (int d = lane; d < D_; d += 32) sum += cp[d] * qp[d];
  for (int off = 16; off; off >>= 1) sum += __shfl_xor(sum, off, 32);
  if (lane == 0) scores[b * S_ + s] = sum;
}

// softmax over S per row b; writes ws copy + attns slice of d_out
__global__ void attn_softmax(const float* __restrict__ scores,
                             float* __restrict__ a,
                             float* __restrict__ attn_out /* + t*S */)
{
  __shared__ float red[S_];
  int b = blockIdx.x, s = threadIdx.x;
  float v = scores[b * S_ + s];
  red[s] = v; __syncthreads();
  for (int o = S_ / 2; o; o >>= 1) { if (s < o) red[s] = fmaxf(red[s], red[s + o]); __syncthreads(); }
  float m = red[0]; __syncthreads();
  float e = expf(v - m);
  red[s] = e; __syncthreads();
  for (int o = S_ / 2; o; o >>= 1) { if (s < o) red[s] += red[s + o]; __syncthreads(); }
  float av = e / red[0];
  a[b * S_ + s] = av;
  attn_out[(size_t)b * (T_ * S_) + s] = av;   // attns[b, t, s]
}

// wctx_bf[b,d] = bf16( sum_s a[b,s] * context[s,b,d] )
__global__ void attn_wctx(const float* __restrict__ a,
                          const float* __restrict__ context,
                          unsigned short* __restrict__ wctx_bf)
{
  int idx = blockIdx.x * blockDim.x + threadIdx.x;  // B*D
  int b = idx >> 10, d = idx & (D_ - 1);
  const float* ab = a + b * S_;
  float sum = 0.f;
  for (int s = 0; s < S_; ++s)
    sum += ab[s] * context[(size_t)s * B_ * D_ + (size_t)b * D_ + d];
  wctx_bf[idx] = f2bf(sum);
}

// ---------------------------------------------------------------------------
extern "C" void kernel_launch(void* const* d_in, const int* in_sizes, int n_in,
                              void* d_out, int out_size, void* d_ws, size_t ws_size,
                              hipStream_t stream) {
  (void)in_sizes; (void)n_in; (void)out_size; (void)ws_size;
  const int*   input   = (const int*)  d_in[0];
  const float* h0_in   = (const float*)d_in[2];
  const float* c0_in   = (const float*)d_in[3];
  const float* context = (const float*)d_in[4];
  const float* init_o  = (const float*)d_in[5];
  const float* lut     = (const float*)d_in[6];
  const float* W_ih0   = (const float*)d_in[7];
  const float* W_hh0   = (const float*)d_in[8];
  const float* b_ih0   = (const float*)d_in[9];
  const float* b_hh0   = (const float*)d_in[10];
  const float* W_ih1   = (const float*)d_in[11];
  const float* W_hh1   = (const float*)d_in[12];
  const float* b_ih1   = (const float*)d_in[13];
  const float* b_hh1   = (const float*)d_in[14];
  const float* W_in    = (const float*)d_in[15];
  const float* W_out   = (const float*)d_in[16];

  // d_out layout: outs[B,T,D], h[2,B,D], c[2,B,D], attns[B,T,S]
  float* outs  = (float*)d_out;
  float* h_st  = outs + (size_t)B_ * T_ * D_;
  float* c_st  = h_st + (size_t)2 * B_ * D_;
  float* attns = c_st + (size_t)2 * B_ * D_;

  // workspace carve (256B aligned)
  char* wp = (char*)d_ws;
  auto carve = [&](size_t bytes) -> void* {
    void* r = (void*)wp; wp += (bytes + 255) & ~(size_t)255; return r;
  };
  unsigned short* Wih0b = (unsigned short*)carve((size_t)G4_ * (E_ + D_) * 2);
  unsigned short* Whh0b = (unsigned short*)carve((size_t)G4_ * D_ * 2);
  unsigned short* Wih1b = (unsigned short*)carve((size_t)G4_ * D_ * 2);
  unsigned short* Whh1b = (unsigned short*)carve((size_t)G4_ * D_ * 2);
  unsigned short* Winb  = (unsigned short*)carve((size_t)D_ * D_ * 2);
  unsigned short* Woutb = (unsigned short*)carve((size_t)D_ * 2 * D_ * 2);
  unsigned short* embb  = (unsigned short*)carve((size_t)T_ * B_ * E_ * 2);
  unsigned short* hbf   = (unsigned short*)carve((size_t)2 * B_ * D_ * 2);
  unsigned short* outbf = (unsigned short*)carve((size_t)B_ * D_ * 2);
  unsigned short* wctxb = (unsigned short*)carve((size_t)B_ * D_ * 2);
  float* gates0 = (float*)carve((size_t)B_ * G4_ * 4);
  float* gates1 = (float*)carve((size_t)B_ * G4_ * 4);
  float* qbuf   = (float*)carve((size_t)B_ * D_ * 4);
  float* scores = (float*)carve((size_t)B_ * S_ * 4);
  float* abuf   = (float*)carve((size_t)B_ * S_ * 4);

  auto cvt = [&](const float* s, unsigned short* d, int n) {
    conv_f32_bf16<<<(n + 255) / 256, 256, 0, stream>>>(s, d, n);
  };
  // one-time per call: bf16 weight/embedding prep
  cvt(W_ih0, Wih0b, G4_ * (E_ + D_));
  cvt(W_hh0, Whh0b, G4_ * D_);
  cvt(W_ih1, Wih1b, G4_ * D_);
  cvt(W_hh1, Whh1b, G4_ * D_);
  cvt(W_in,  Winb,  D_ * D_);
  cvt(W_out, Woutb, D_ * 2 * D_);
  embed_bf16<<<(T_ * B_ * E_) / 256, 256, 0, stream>>>(input, lut, embb);
  cvt(h0_in, hbf, 2 * B_ * D_);
  cvt(init_o, outbf, B_ * D_);
  hipMemcpyAsync(h_st, h0_in, (size_t)2 * B_ * D_ * 4, hipMemcpyDeviceToDevice, stream);
  hipMemcpyAsync(c_st, c0_in, (size_t)2 * B_ * D_ * 4, hipMemcpyDeviceToDevice, stream);

  unsigned short* h0b = hbf;
  unsigned short* h1b = hbf + (size_t)B_ * D_;
  float* h0s = h_st;                 float* h1s = h_st + (size_t)B_ * D_;
  float* c0s = c_st;                 float* c1s = c_st + (size_t)B_ * D_;

  // each wave covers 64x16; 4 waves per block -> one block per 64 N columns
  const int gatesBlocks = G4_ / 64;                     // 64
  const int dBlocks     = D_  / 64;                     // 16
  const int pwBlocks    = (B_ * D_) / 256;              // 256

  for (int t = 0; t < T_; ++t) {
    // layer 0 gates: emb_t | out_prev | h0_prev vs W_ih0[:, :E] / W_ih0[:, E:] / W_hh0
    gemm_bf16_wmma<<<gatesBlocks, 128, 0, stream>>>(
        gates0, G4_, nullptr, 0,
        embb + (size_t)t * B_ * E_, E_,      Wih0b,        E_ + D_, E_,
        outbf,                      D_,      Wih0b + E_,   E_ + D_, D_,
        h0b,                        D_,      Whh0b,        D_,      D_,
        b_ih0, b_hh0, 0);
    lstm_pointwise<<<pwBlocks, 256, 0, stream>>>(gates0, c0s, h0s, h0b);

    // layer 1 gates: h0_new | h1_prev
    gemm_bf16_wmma<<<gatesBlocks, 128, 0, stream>>>(
        gates1, G4_, nullptr, 0,
        h0b, D_, Wih1b, D_, D_,
        h1b, D_, Whh1b, D_, D_,
        nullptr, 0, nullptr, 0, 0,
        b_ih1, b_hh1, 0);
    lstm_pointwise<<<pwBlocks, 256, 0, stream>>>(gates1, c1s, h1s, h1b);

    // attention: q = h1 @ W_in^T
    gemm_bf16_wmma<<<dBlocks, 128, 0, stream>>>(
        qbuf, D_, nullptr, 0,
        h1b, D_, Winb, D_, D_,
        nullptr, 0, nullptr, 0, 0,
        nullptr, 0, nullptr, 0, 0,
        nullptr, nullptr, 0);
    attn_scores<<<(B_ * S_) / 4, 128, 0, stream>>>(qbuf, context, scores);
    attn_softmax<<<B_, S_, 0, stream>>>(scores, abuf, attns + (size_t)t * S_);
    attn_wctx<<<pwBlocks, 256, 0, stream>>>(abuf, context, wctxb);

    // out = tanh([wctx, h1] @ W_out^T) -> outs[:, t, :] and bf16 feed for t+1
    gemm_bf16_wmma<<<dBlocks, 128, 0, stream>>>(
        outs + (size_t)t * D_, T_ * D_, outbf, D_,
        wctxb, D_, Woutb,       2 * D_, D_,
        h1b,   D_, Woutb + D_,  2 * D_, D_,
        nullptr, 0, nullptr, 0, 0,
        nullptr, nullptr, 1);
  }
}
